// Encoder_13666585936394
// MI455X (gfx1250) — compile-verified, matrix operand order checked by reading, not run on hardware
//
#include <hip/hip_runtime.h>

// ---------------------------------------------------------------------------
// Problem constants (match reference)
// ---------------------------------------------------------------------------
static constexpr int BB    = 4;
static constexpr int SS    = 1024;
static constexpr int DIM   = 1024;
static constexpr int HEADS = 16;
static constexpr int NLAY  = 6;
static constexpr int FFN   = 4096;    // 4*DIM

// ---------------------------------------------------------------------------
// CDNA5 WMMA plumbing (wave32, 16x16x32 bf16 -> f32)
// ---------------------------------------------------------------------------
typedef __bf16         bf16x16 __attribute__((ext_vector_type(16)));
typedef float          f32x8   __attribute__((ext_vector_type(8)));
typedef unsigned short u16x8   __attribute__((ext_vector_type(8)));
typedef unsigned short u16x4   __attribute__((ext_vector_type(4)));
typedef unsigned int   u32x4   __attribute__((ext_vector_type(4)));
typedef int            i32x4   __attribute__((ext_vector_type(4)));
typedef int            i32x8   __attribute__((ext_vector_type(8)));

__device__ __forceinline__ f32x8 wmma_bf16(bf16x16 a, bf16x16 b, f32x8 c) {
  return __builtin_amdgcn_wmma_f32_16x16x32_bf16(
      /*neg_a=*/false, a, /*neg_b=*/false, b,
      /*c_mod=*/(short)0, c, /*reuse_a=*/false, /*reuse_b=*/false);
}

__device__ __forceinline__ unsigned short f2bf(float f) {
  unsigned int u = __float_as_uint(f);
  u = (u + 0x7FFFu + ((u >> 16) & 1u)) >> 16;   // round-to-nearest-even
  return (unsigned short)u;
}

// Load one 16x32 bf16 fragment (A-layout; B fragments load identically from a
// transposed LDS tile).  Per ISA 7.12.2:
//   lanes 0-15 : row M=lane,    K in [0..7] then [16..23]
//   lanes 16-31: row M=lane-16, K in [8..15] then [24..31]
// Requires 16B-aligned rows (ld*2 % 16 == 0).
__device__ __forceinline__ bf16x16 lds_frag(const unsigned short* base, int ld, int lane) {
  int row = lane & 15;
  int k0  = (lane >> 4) << 3;                   // 0 or 8
  const unsigned short* p = base + row * ld;
  union { u16x8 h[2]; bf16x16 v; } u;
  u.h[0] = *(const u16x8*)(p + k0);
  u.h[1] = *(const u16x8*)(p + k0 + 16);
  return u.v;
}

static constexpr int LDT = 40;                   // padded LDS row (ushorts) = 80B

// ---------------------------------------------------------------------------
// Tensor Data Mover: 2D tile load (bf16 elements, K-contiguous rows) into LDS
// with padding so the LDS row stride is 64B + 16B = 80B (matches LDT).
// D# layout per CDNA5 ISA 8.3/8.4; issued by one wave, EXEC ignored.
// ---------------------------------------------------------------------------
#if defined(__has_builtin)
#if __has_builtin(__builtin_amdgcn_tensor_load_to_lds)
#define HAVE_TDM 1
#endif
#endif
#ifndef HAVE_TDM
#define HAVE_TDM 0
#endif

#if HAVE_TDM
__device__ __forceinline__ void tdm_load_tile_2d(const unsigned short* gbase,
                                                 unsigned lds_byte_addr,
                                                 unsigned row_len_elems,   // K
                                                 unsigned n_rows_tensor,   // rows in tensor
                                                 unsigned tile_rows) {     // rows in tile
  unsigned long long ga = (unsigned long long)(uintptr_t)gbase;
  u32x4 g0;
  g0[0] = 0x1u;                                       // count=1, user-mode descriptor
  g0[1] = lds_byte_addr;                              // lds_addr
  g0[2] = (unsigned)(ga & 0xFFFFFFFFu);               // global_addr[31:0]
  g0[3] = ((unsigned)(ga >> 32) & 0x01FFFFFFu)        // global_addr[56:32]
        | (2u << 30);                                 // type = 2 ("image")
  i32x8 g1;
  // workgroup_mask=0 | data_size=1(2B) | pad_enable | pad_interval=3(16 dw=64B)
  // | pad_amount=3(4 dw=16B)
  g1[0] = (1 << 16) | (1 << 20) | (3 << 22) | (3 << 25);
  g1[1] = (int)((row_len_elems & 0xFFFFu) << 16);     // tensor_dim0[15:0] (atomic addr=0)
  g1[2] = (int)((row_len_elems >> 16) | ((n_rows_tensor & 0xFFFFu) << 16));
  g1[3] = (int)((n_rows_tensor >> 16) | (32u << 16)); // tile_dim0 = 32 elems
  g1[4] = (int)(tile_rows & 0xFFFFu);                 // tile_dim1 (tile_dim2 = 0)
  g1[5] = (int)row_len_elems;                         // tensor_dim0_stride (48b, low word)
  g1[6] = 0;
  g1[7] = 0;
  i32x4 z4 = {0, 0, 0, 0};
#if __clang_major__ >= 23
  i32x8 z8 = {0, 0, 0, 0, 0, 0, 0, 0};
  __builtin_amdgcn_tensor_load_to_lds(g0, g1, z4, z4, z8, 0);
#else
  __builtin_amdgcn_tensor_load_to_lds(g0, g1, z4, z4, 0);
#endif
}
#endif  // HAVE_TDM

// ---------------------------------------------------------------------------
// Embedding + sinusoidal positional encoding
// ---------------------------------------------------------------------------
__global__ void __launch_bounds__(256)
embed_pe_kernel(const int* __restrict__ ids, const int* __restrict__ pos,
                const float* __restrict__ emb, float* __restrict__ xs) {
  int tok = blockIdx.x;
  int id  = ids[tok];
  float p = (float)pos[tok];
  const float* e = emb + (size_t)id * DIM;
  float* o = xs + (size_t)tok * DIM;
  for (int d = threadIdx.x; d < DIM; d += 256) {
    int i = d >> 1;
    float freq = __expf(-9.210340372f * (2.0f * (float)i) * (1.0f / (float)DIM));
    float ang  = p * freq;
    float pe   = (d & 1) ? __cosf(ang) : __sinf(ang);
    o[d] = e[d] + pe;
  }
}

// ---------------------------------------------------------------------------
// f32 -> bf16 conversion (one float4 group per thread)
// ---------------------------------------------------------------------------
__global__ void __launch_bounds__(256)
convert_bf16_kernel(const float* __restrict__ x, unsigned short* __restrict__ y, int n4) {
  int i = blockIdx.x * 256 + threadIdx.x;
  if (i < n4) {
    float4 v = ((const float4*)x)[i];
    u16x4 o;
    o[0] = f2bf(v.x); o[1] = f2bf(v.y); o[2] = f2bf(v.z); o[3] = f2bf(v.w);
    *(u16x4*)(y + (size_t)i * 4) = o;
  }
}

// ---------------------------------------------------------------------------
// Weight transpose + bf16 convert:  W[K][N] f32  ->  Wt[N][K] bf16
// ---------------------------------------------------------------------------
__global__ void __launch_bounds__(256)
wtrans_bf16_kernel(const float* __restrict__ W, unsigned short* __restrict__ Wt,
                   int K, int N) {
  __shared__ float tile[32][33];
  int k0 = blockIdx.y * 32, n0 = blockIdx.x * 32;
  int tx = threadIdx.x & 31, ty = threadIdx.x >> 5;      // 32 x 8
  for (int r = ty; r < 32; r += 8)
    tile[r][tx] = W[(size_t)(k0 + r) * N + n0 + tx];
  __syncthreads();
  for (int r = ty; r < 32; r += 8)
    Wt[(size_t)(n0 + r) * K + k0 + tx] = f2bf(tile[tx][r]);
}

// ---------------------------------------------------------------------------
// WMMA GEMM:  C[M,N] = op( A[M,K] @ W[K,N] + bias + residual )
// A is bf16 row-major [M][K]; W is bf16 pre-transposed Wt[N][K].
// Block = 128 threads (4 waves); tile 128(M) x 64(N), K-step 32.
// Each wave computes 32x64 -> 2 A-frags x 4 B-frags -> 8 back-to-back WMMAs.
//
// Double-buffered TDM pipeline: wave 0 issues the *next* K-tile's two
// tensor_load_to_lds into the ping-pong buffer, then waits TENSORcnt<=2
// (in-order completion => current tile's DMA is done) so the current tile's
// WMMA burst overlaps the next tile's DMA.  Fallback = b128 manual staging.
// ---------------------------------------------------------------------------
static constexpr int TMG = 128, TN = 64, TK = 32;

__global__ void __launch_bounds__(128)
gemm_wmma_kernel(const unsigned short* __restrict__ A,
                 const unsigned short* __restrict__ Wt,
                 const float* __restrict__ bias, const float* __restrict__ resid,
                 float* __restrict__ C, int M, int N, int K, int do_relu) {
  __shared__ unsigned short As[2][TMG][LDT];     // [buf][m][k]
  __shared__ unsigned short Bs[2][TN][LDT];      // [buf][n][k]

  const int tid  = threadIdx.x;
  const int lane = tid & 31;
  const int wave = tid >> 5;
  const int m0   = blockIdx.y * TMG;
  const int n0   = blockIdx.x * TN;

  f32x8 acc[2][4] = {};

#if HAVE_TDM
  const unsigned as_base = (unsigned)(uintptr_t)&As[0][0][0];
  const unsigned bs_base = (unsigned)(uintptr_t)&Bs[0][0][0];
  const unsigned as_sz   = (unsigned)(TMG * LDT * 2);
  const unsigned bs_sz   = (unsigned)(TN * LDT * 2);
  // Prologue: DMA tile 0 into buffer 0.
  if (wave == 0) {
    tdm_load_tile_2d(A  + (size_t)m0 * K, as_base, (unsigned)K, (unsigned)M, TMG);
    tdm_load_tile_2d(Wt + (size_t)n0 * K, bs_base, (unsigned)K, (unsigned)N, TN);
  }
#endif

  for (int k0 = 0; k0 < K; k0 += TK) {
    const int cb = (k0 / TK) & 1;
#if HAVE_TDM
    if (wave == 0) {
      if (k0 + TK < K) {
        const int nb = cb ^ 1;
        tdm_load_tile_2d(A  + (size_t)m0 * K + k0 + TK, as_base + nb * as_sz,
                         (unsigned)K, (unsigned)M, TMG);
        tdm_load_tile_2d(Wt + (size_t)n0 * K + k0 + TK, bs_base + nb * bs_sz,
                         (unsigned)K, (unsigned)N, TN);
        // 2 in-flight loads belong to the next tile; current tile is complete.
        __builtin_amdgcn_s_wait_tensorcnt(2);
      } else {
        __builtin_amdgcn_s_wait_tensorcnt(0);
      }
    }
    __syncthreads();                             // current buffer ready for all waves
#else
    for (int i = tid; i < (TMG * TK) / 8; i += 128) {      // 16B chunks
      int row = i >> 2, c8 = i & 3;
      *(u16x8*)&As[cb][row][c8 * 8] =
          *(const u16x8*)(A + (size_t)(m0 + row) * K + k0 + c8 * 8);
    }
    for (int i = tid; i < (TN * TK) / 8; i += 128) {
      int row = i >> 2, c8 = i & 3;
      *(u16x8*)&Bs[cb][row][c8 * 8] =
          *(const u16x8*)(Wt + (size_t)(n0 + row) * K + k0 + c8 * 8);
    }
    __syncthreads();
#endif
    // Load all fragments first, then fire 8 WMMAs back-to-back.
    bf16x16 a0 = lds_frag(&As[cb][wave * 32 + 0][0],  LDT, lane);
    bf16x16 a1 = lds_frag(&As[cb][wave * 32 + 16][0], LDT, lane);
    bf16x16 b0 = lds_frag(&Bs[cb][0][0],  LDT, lane);
    bf16x16 b1 = lds_frag(&Bs[cb][16][0], LDT, lane);
    bf16x16 b2 = lds_frag(&Bs[cb][32][0], LDT, lane);
    bf16x16 b3 = lds_frag(&Bs[cb][48][0], LDT, lane);
    acc[0][0] = wmma_bf16(a0, b0, acc[0][0]);
    acc[0][1] = wmma_bf16(a0, b1, acc[0][1]);
    acc[0][2] = wmma_bf16(a0, b2, acc[0][2]);
    acc[0][3] = wmma_bf16(a0, b3, acc[0][3]);
    acc[1][0] = wmma_bf16(a1, b0, acc[1][0]);
    acc[1][1] = wmma_bf16(a1, b1, acc[1][1]);
    acc[1][2] = wmma_bf16(a1, b2, acc[1][2]);
    acc[1][3] = wmma_bf16(a1, b3, acc[1][3]);
    __syncthreads();   // all waves done reading buf cb before tile k0+2*TK targets it
  }

  // Epilogue: C-fragment layout = col (lane&15), rows r + 8*(lane>=16).
  const int col = lane & 15;
  const int hi  = lane >> 4;
#pragma unroll
  for (int a = 0; a < 2; ++a) {
#pragma unroll
    for (int j = 0; j < 4; ++j) {
      int n = n0 + j * 16 + col;
      float bv = bias ? bias[n] : 0.0f;
#pragma unroll
      for (int r = 0; r < 8; ++r) {
        int m = m0 + wave * 32 + a * 16 + r + hi * 8;
        float v = acc[a][j][r] + bv;
        if (resid) v += resid[(size_t)m * N + n];
        if (do_relu) v = fmaxf(v, 0.0f);
        C[(size_t)m * N + n] = v;
      }
    }
  }
}

// ---------------------------------------------------------------------------
// Flash attention.  qkv: (B,S,3*DIM) f32 with per-head layout d = dk*H + h.
// Block: 4 waves, 64 query rows (16 per wave), streaming 32-key chunks.
// ---------------------------------------------------------------------------
static constexpr int LDQ = 72;                   // 64 dk + 8 pad

__global__ void __launch_bounds__(128)
attn_flash_kernel(const float* __restrict__ qkv, const int* __restrict__ mask,
                  float* __restrict__ out) {
  const int bh = blockIdx.y;
  const int b  = bh / HEADS, h = bh % HEADS;
  const int q0 = blockIdx.x * 64;
  const int tid = threadIdx.x, lane = tid & 31, wave = tid >> 5;
  const int col = lane & 15, hi = lane >> 4;
  const size_t rs = (size_t)3 * DIM;

  __shared__ unsigned short Qs[64][LDQ];         // [q][dk]
  __shared__ unsigned short Ks[32][LDQ];         // [key][dk]
  __shared__ unsigned short Vt[64][LDT];         // [dk][key] (transposed)
  __shared__ unsigned short Ps[4][16][LDT];      // per-wave probs [q][key]

  for (int i = tid; i < 64 * 64; i += 128) {
    int qr = i >> 6, d = i & 63;
    Qs[qr][d] = f2bf(qkv[((size_t)(b * SS + q0 + qr)) * rs + d * HEADS + h]);
  }
  __syncthreads();
  bf16x16 qf0 = lds_frag(&Qs[wave * 16][0],  LDQ, lane);
  bf16x16 qf1 = lds_frag(&Qs[wave * 16][32], LDQ, lane);

  float m_run[8], l_run[8];
  f32x8 o[4] = {};
#pragma unroll
  for (int r = 0; r < 8; ++r) { m_run[r] = -3.0e38f; l_run[r] = 0.0f; }

  for (int kb = 0; kb < SS; kb += 32) {
    __syncthreads();
    for (int i = tid; i < 32 * 64; i += 128) {
      int kr = i >> 6, d = i & 63;
      size_t base = ((size_t)(b * SS + kb + kr)) * rs + d * HEADS + h;
      Ks[kr][d] = f2bf(qkv[base + DIM]);
      Vt[d][kr] = f2bf(qkv[base + 2 * DIM]);
    }
    __syncthreads();

    f32x8 s0 = {}, s1 = {};
    s0 = wmma_bf16(qf0, lds_frag(&Ks[0][0],   LDQ, lane), s0);
    s0 = wmma_bf16(qf1, lds_frag(&Ks[0][32],  LDQ, lane), s0);
    s1 = wmma_bf16(qf0, lds_frag(&Ks[16][0],  LDQ, lane), s1);
    s1 = wmma_bf16(qf1, lds_frag(&Ks[16][32], LDQ, lane), s1);

    const int key0 = kb + col, key1 = key0 + 16;
#pragma unroll
    for (int r = 0; r < 8; ++r) {
      int q = q0 + wave * 16 + r + hi * 8;
      const size_t mrow = ((size_t)b * SS + q) * SS;
      float a0 = (mask[mrow + key0] == 0) ? -1.0e10f : 0.0f;
      float a1 = (mask[mrow + key1] == 0) ? -1.0e10f : 0.0f;
      s0[r] = s0[r] * 0.125f + a0;               // 1/sqrt(64)
      s1[r] = s1[r] * 0.125f + a1;
    }

#pragma unroll
    for (int r = 0; r < 8; ++r) {
      float v = fmaxf(s0[r], s1[r]);
#pragma unroll
      for (int off = 1; off < 16; off <<= 1) v = fmaxf(v, __shfl_xor(v, off, 32));
      float mnew  = fmaxf(m_run[r], v);
      float alpha = __expf(m_run[r] - mnew);
      m_run[r] = mnew;
      float p0 = __expf(s0[r] - mnew);
      float p1 = __expf(s1[r] - mnew);
      float sum = p0 + p1;
#pragma unroll
      for (int off = 1; off < 16; off <<= 1) sum += __shfl_xor(sum, off, 32);
      l_run[r] = l_run[r] * alpha + sum;
#pragma unroll
      for (int j = 0; j < 4; ++j) o[j][r] *= alpha;
      int prow = r + hi * 8;
      Ps[wave][prow][col]      = f2bf(p0);
      Ps[wave][prow][col + 16] = f2bf(p1);
    }

    bf16x16 pf = lds_frag(&Ps[wave][0][0], LDT, lane);
#pragma unroll
    for (int j = 0; j < 4; ++j)
      o[j] = wmma_bf16(pf, lds_frag(&Vt[j * 16][0], LDT, lane), o[j]);
  }

#pragma unroll
  for (int j = 0; j < 4; ++j) {
    int dloc = j * 16 + col;
#pragma unroll
    for (int r = 0; r < 8; ++r) {
      int q = q0 + wave * 16 + r + hi * 8;
      out[((size_t)(b * SS + q)) * DIM + dloc * HEADS + h] = o[j][r] / l_run[r];
    }
  }
}

// ---------------------------------------------------------------------------
// LayerNorm over last dim (row-per-block)
// ---------------------------------------------------------------------------
__global__ void __launch_bounds__(256)
layernorm_kernel(const float* __restrict__ x, const float* __restrict__ g,
                 const float* __restrict__ be, float* __restrict__ y) {
  __shared__ float red[2][8];
  const int row = blockIdx.x;
  const float* xr = x + (size_t)row * DIM;
  float s = 0.0f, s2 = 0.0f;
  for (int i = threadIdx.x; i < DIM; i += 256) { float v = xr[i]; s += v; s2 += v * v; }
#pragma unroll
  for (int off = 1; off < 32; off <<= 1) { s += __shfl_xor(s, off, 32); s2 += __shfl_xor(s2, off, 32); }
  int wv = threadIdx.x >> 5, ln = threadIdx.x & 31;
  if (ln == 0) { red[0][wv] = s; red[1][wv] = s2; }
  __syncthreads();
  float ts = 0.0f, ts2 = 0.0f;
#pragma unroll
  for (int w = 0; w < 8; ++w) { ts += red[0][w]; ts2 += red[1][w]; }
  float mu   = ts / (float)DIM;
  float var  = ts2 / (float)DIM - mu * mu;
  float rstd = rsqrtf(var + 1e-5f);
  float* yr = y + (size_t)row * DIM;
  for (int i = threadIdx.x; i < DIM; i += 256)
    yr[i] = (xr[i] - mu) * rstd * g[i] + be[i];
}

// ---------------------------------------------------------------------------
// Host-side orchestration
// ---------------------------------------------------------------------------
static inline void run_gemm(const unsigned short* A, const unsigned short* Wt,
                            const float* bias, const float* resid, float* C,
                            int M, int N, int K, int do_relu, hipStream_t stream) {
  gemm_wmma_kernel<<<dim3(N / TN, M / TMG), 128, 0, stream>>>(
      A, Wt, bias, resid, C, M, N, K, do_relu);
}

extern "C" void kernel_launch(void* const* d_in, const int* in_sizes, int n_in,
                              void* d_out, int out_size, void* d_ws, size_t ws_size,
                              hipStream_t stream) {
  (void)in_sizes; (void)n_in; (void)out_size; (void)ws_size;
  const int*   ids   = (const int*)d_in[0];
  const int*   pos   = (const int*)d_in[1];
  const int*   mask  = (const int*)d_in[2];
  const float* emb   = (const float*)d_in[3];
  const float* w_qkv = (const float*)d_in[4];
  const float* w_out = (const float*)d_in[5];
  const float* w1    = (const float*)d_in[6];
  const float* b1    = (const float*)d_in[7];
  const float* w2    = (const float*)d_in[8];
  const float* b2    = (const float*)d_in[9];
  const float* g1    = (const float*)d_in[10];
  const float* be1   = (const float*)d_in[11];
  const float* g2    = (const float*)d_in[12];
  const float* be2   = (const float*)d_in[13];

  const size_t T = (size_t)BB * SS;              // 4096 tokens
  float* ws  = (float*)d_ws;
  float* xsA = ws;                               // T*DIM
  float* xsB = xsA + T * DIM;                    // T*DIM
  float* qkv = xsB + T * DIM;                    // T*3*DIM
  float* atn = qkv + T * 3 * DIM;                // T*DIM
  float* tmp = atn + T * DIM;                    // T*DIM
  float* ln1 = tmp + T * DIM;                    // T*DIM
  float* ffn = ln1 + T * DIM;                    // T*FFN
  unsigned short* Abf = (unsigned short*)(ffn + T * FFN);   // T*FFN bf16
  unsigned short* Wtb = Abf + T * FFN;                      // FFN*DIM bf16

  embed_pe_kernel<<<(int)T, 256, 0, stream>>>(ids, pos, emb, xsA);

  float* cur = xsA;
  float* nxt = xsB;
  for (int l = 0; l < NLAY; ++l) {
    // ---- QKV projection: (T,DIM) @ (DIM,3*DIM) ----
    convert_bf16_kernel<<<(int)(T * DIM / 4 / 256), 256, 0, stream>>>(cur, Abf, (int)(T * DIM / 4));
    wtrans_bf16_kernel<<<dim3(3 * DIM / 32, DIM / 32), 256, 0, stream>>>(
        w_qkv + (size_t)l * DIM * 3 * DIM, Wtb, DIM, 3 * DIM);
    run_gemm(Abf, Wtb, nullptr, nullptr, qkv, (int)T, 3 * DIM, DIM, 0, stream);

    // ---- Flash attention ----
    attn_flash_kernel<<<dim3(SS / 64, BB * HEADS), 128, 0, stream>>>(qkv, mask, atn);

    // ---- Output projection + residual ----
    convert_bf16_kernel<<<(int)(T * DIM / 4 / 256), 256, 0, stream>>>(atn, Abf, (int)(T * DIM / 4));
    wtrans_bf16_kernel<<<dim3(DIM / 32, DIM / 32), 256, 0, stream>>>(
        w_out + (size_t)l * DIM * DIM, Wtb, DIM, DIM);
    run_gemm(Abf, Wtb, nullptr, cur, tmp, (int)T, DIM, DIM, 0, stream);
    layernorm_kernel<<<(int)T, 256, 0, stream>>>(tmp, g1 + l * DIM, be1 + l * DIM, ln1);

    // ---- FFN up (+bias, ReLU) ----
    convert_bf16_kernel<<<(int)(T * DIM / 4 / 256), 256, 0, stream>>>(ln1, Abf, (int)(T * DIM / 4));
    wtrans_bf16_kernel<<<dim3(FFN / 32, DIM / 32), 256, 0, stream>>>(
        w1 + (size_t)l * DIM * FFN, Wtb, DIM, FFN);
    run_gemm(Abf, Wtb, b1 + (size_t)l * FFN, nullptr, ffn, (int)T, FFN, DIM, 1, stream);

    // ---- FFN down (+bias, residual) ----
    convert_bf16_kernel<<<(int)(T * FFN / 4 / 256), 256, 0, stream>>>(ffn, Abf, (int)(T * FFN / 4));
    wtrans_bf16_kernel<<<dim3(DIM / 32, FFN / 32), 256, 0, stream>>>(
        w2 + (size_t)l * FFN * DIM, Wtb, FFN, DIM);
    run_gemm(Abf, Wtb, b2 + (size_t)l * DIM, ln1, tmp, (int)T, DIM, FFN, 0, stream);

    float* outp = (l == NLAY - 1) ? (float*)d_out : nxt;
    layernorm_kernel<<<(int)T, 256, 0, stream>>>(tmp, g2 + l * DIM, be2 + l * DIM, outp);
    cur = outp;
    nxt = (cur == xsA) ? xsB : xsA;
  }
}
